// GPT2LMHeadModelWrapper_27135603376468
// MI455X (gfx1250) — compile-verified, matrix operand order checked by reading, not run on hardware
//
#include <hip/hip_runtime.h>
#include <cstdint>
#include <cstddef>

// ---------------- model constants ----------------
#define LCNT   12
#define HCNT   12
#define HDIM   64
#define DMODEL 768
#define FFDIM  3072
#define VOCAB  50257
#define SEQ    1024
#define BATCH  2
#define MTOK   2048                       // BATCH*SEQ
static const size_t LOGITS_ELEMS = (size_t)MTOK * VOCAB;
static const size_t KVL = (size_t)BATCH * HCNT * SEQ * HDIM;  // one k (or v) tensor

typedef unsigned short u16;
typedef __attribute__((ext_vector_type(16))) __bf16 v16bf;
typedef __attribute__((ext_vector_type(8)))  float  v8f;

// ---------------- helpers ----------------
__device__ __forceinline__ u16 f2bf(float f) {
    unsigned u = __builtin_bit_cast(unsigned, f);
    u += 0x7fffu + ((u >> 16) & 1u);      // round-to-nearest-even
    return (u16)(u >> 16);
}

struct B32x2 { uint4 lo, hi; };
__device__ __forceinline__ v16bf load_frag(const u16* p0, const u16* p1) {
    B32x2 t;
    t.lo = *(const uint4*)p0;
    t.hi = *(const uint4*)p1;
    return __builtin_bit_cast(v16bf, t);
}

__device__ __forceinline__ v8f wmma_bf16(v16bf a, v16bf b, v8f c) {
    // D = A(16x32 bf16) x B(32x16 bf16) + C(16x16 f32)
    return __builtin_amdgcn_wmma_f32_16x16x32_bf16(false, a, false, b, (short)0, c,
                                                   false, false);
}

__device__ __forceinline__ float gelu_tanh(float x) {
    float c = 0.7978845608028654f * (x + 0.044715f * x * x * x);
    return 0.5f * x * (1.0f + tanhf(c));
}

// CDNA5 async memory->LDS copy: 16B per lane, tracked by ASYNCcnt.
// LDS operand is a VGPR holding the LDS byte address; for shared-aperture flat
// pointers the low 32 address bits ARE the LDS address (ISA 10.2).
__device__ __forceinline__ void async_b128(void* lds, const void* g) {
    unsigned l = (unsigned)(size_t)lds;
    asm volatile("global_load_async_to_lds_b128 %0, %1, off"
                 : : "v"(l), "v"(g) : "memory");
}
// Wait until at most N async ops remain outstanding (in-order completion).
template <int N>
__device__ __forceinline__ void wait_async() {
    asm volatile("s_wait_asynccnt %0" : : "i"(N) : "memory");
}

// ---------------- weight convert / transpose ----------------
// in: fp32 [z][K][N] row-major  ->  out: bf16 [z][N][K]
__global__ __launch_bounds__(256) void k_tconv(const float* __restrict__ in,
                                               u16* __restrict__ out, int K, int N) {
    __shared__ float t[32][33];
    int kb = blockIdx.y * 32, nb = blockIdx.x * 32;
    const float* I = in + (size_t)blockIdx.z * K * N;
    u16* O = out + (size_t)blockIdx.z * K * N;
    int tx = threadIdx.x, ty = threadIdx.y;          // 32 x 8
    #pragma unroll
    for (int j = 0; j < 32; j += 8)
        t[ty + j][tx] = I[(size_t)(kb + ty + j) * N + nb + tx];
    __syncthreads();
    #pragma unroll
    for (int j = 0; j < 32; j += 8)
        O[(size_t)(nb + ty + j) * K + kb + tx] = f2bf(t[tx][ty + j]);
}

// plain elementwise fp32 -> bf16 (wte: already [V, D] = [N, K])
__global__ __launch_bounds__(256) void k_conv(const float* __restrict__ in,
                                              u16* __restrict__ out, size_t n) {
    size_t i = (size_t)blockIdx.x * 256 + threadIdx.x;
    size_t stride = (size_t)gridDim.x * 256;
    for (; i < n; i += stride) out[i] = f2bf(in[i]);
}

// ---------------- embedding ----------------
__global__ __launch_bounds__(256) void k_embed(const int* __restrict__ ids,
                                               const int* __restrict__ cp,
                                               const float* __restrict__ wte,
                                               const float* __restrict__ wpe,
                                               float* __restrict__ h) {
    size_t i = (size_t)blockIdx.x * 256 + threadIdx.x;
    if (i >= (size_t)MTOK * DMODEL) return;
    int t = (int)(i / DMODEL), d = (int)(i % DMODEL);
    int s = t & (SEQ - 1);
    int pos = s + cp[0];
    h[i] = wte[(size_t)ids[t] * DMODEL + d] + wpe[(size_t)pos * DMODEL + d];
}

// ---------------- layernorm (fp32 in -> bf16 out) ----------------
__global__ __launch_bounds__(256) void k_ln(const float* __restrict__ x,
                                            const float* __restrict__ g,
                                            const float* __restrict__ bta,
                                            u16* __restrict__ y) {
    __shared__ float red[256];
    int tid = threadIdx.x;
    const float* xr = x + (size_t)blockIdx.x * DMODEL;
    float v0 = xr[tid], v1 = xr[tid + 256], v2 = xr[tid + 512];
    red[tid] = v0 + v1 + v2;
    __syncthreads();
    #pragma unroll
    for (int off = 128; off; off >>= 1) {
        if (tid < off) red[tid] += red[tid + off];
        __syncthreads();
    }
    float mean = red[0] * (1.0f / DMODEL);
    __syncthreads();
    float d0 = v0 - mean, d1 = v1 - mean, d2 = v2 - mean;
    red[tid] = d0 * d0 + d1 * d1 + d2 * d2;
    __syncthreads();
    #pragma unroll
    for (int off = 128; off; off >>= 1) {
        if (tid < off) red[tid] += red[tid + off];
        __syncthreads();
    }
    float r = rsqrtf(red[0] * (1.0f / DMODEL) + 1e-5f);
    u16* yr = y + (size_t)blockIdx.x * DMODEL;
    yr[tid]       = f2bf(d0 * r * g[tid]       + bta[tid]);
    yr[tid + 256] = f2bf(d1 * r * g[tid + 256] + bta[tid + 256]);
    yr[tid + 512] = f2bf(d2 * r * g[tid + 512] + bta[tid + 512]);
}

// ---------------- WMMA GEMM ----------------
// C[M,N] = act( A[M,K](bf16) @ W[N,K](bf16)^T + bias ) (+res); optional bf16 copy.
// Block 256 thr = 8 waves; tile 128x128x32; wave = 32(M) x 64(N) = 2x4 subtiles.
// Tiles are staged with a 2-deep global_load_async_to_lds_b128 pipeline:
// issue next tile's DMA, s_wait_asynccnt 4 (current tile done, in-order),
// then WMMA on the current LDS buffer while the DMA engine fills the other.
#define BK  32
#define LDT 40   // padded LDS row stride in u16 (80B -> keeps 16B align, spreads banks)

template <bool GELU, bool RES, bool WB16, bool BIAS, bool NGUARD>
__global__ __launch_bounds__(256) void k_gemm(const u16* __restrict__ A,
                                              const u16* __restrict__ W,
                                              const float* __restrict__ bias,
                                              const float* res, float* C,
                                              u16* __restrict__ C16,
                                              int M, int N, int K) {
    __shared__ u16 As[2][128 * LDT];
    __shared__ u16 Ws[2][128 * LDT];
    int tid = threadIdx.x;
    int lane = tid & 31, wave = tid >> 5;
    int wm = (wave & 3) * 32, wn = (wave >> 2) * 64;
    int rl = lane & 15, half = lane >> 4;
    int m0 = blockIdx.y * 128, n0 = blockIdx.x * 128;

    v8f zero = {};
    v8f acc[2][4];
    #pragma unroll
    for (int i = 0; i < 2; ++i)
        #pragma unroll
        for (int j = 0; j < 4; ++j) acc[i][j] = zero;

    int lr = tid >> 1;               // 0..127 (row of tile loaded by this thread)
    int lc = (tid & 1) * 16;         // 0 or 16 (u16 elements)
    const u16* Ap = A + (size_t)(m0 + lr) * K + lc;
    int wrow = n0 + lr;
    // Out-of-range weight rows (lm_head tail tile) clamp to row 0: they only
    // feed accumulator columns n>=N which the epilogue never stores.
    bool wok = (!NGUARD) || (wrow < N);
    const u16* Wp = W + (size_t)(wok ? wrow : 0) * K + lc;
    int ldso = lr * LDT + lc;        // this thread's LDS staging offset

    // prologue: kick off DMA for k-step 0 into buffer 0
    async_b128(&As[0][ldso],     Ap);
    async_b128(&As[0][ldso + 8], Ap + 8);
    async_b128(&Ws[0][ldso],     Wp);
    async_b128(&Ws[0][ldso + 8], Wp + 8);

    int buf = 0;
    for (int k0 = 0; k0 < K; k0 += BK) {
        int nb = buf ^ 1;
        if (k0 + BK < K) {
            // issue next tile's DMA into the other buffer
            async_b128(&As[nb][ldso],     Ap + k0 + BK);
            async_b128(&As[nb][ldso + 8], Ap + k0 + BK + 8);
            async_b128(&Ws[nb][ldso],     Wp + k0 + BK);
            async_b128(&Ws[nb][ldso + 8], Wp + k0 + BK + 8);
            wait_async<4>();   // <=4 outstanding: current tile's 4 are complete
        } else {
            wait_async<0>();
        }
        __syncthreads();       // current buffer visible to all waves

        v16bf af[2], wf[4];
        #pragma unroll
        for (int i = 0; i < 2; ++i) {
            const u16* p = &As[buf][(wm + i * 16 + rl) * LDT + half * 8];
            af[i] = load_frag(p, p + 16);
        }
        #pragma unroll
        for (int j = 0; j < 4; ++j) {
            const u16* p = &Ws[buf][(wn + j * 16 + rl) * LDT + half * 8];
            wf[j] = load_frag(p, p + 16);
        }
        #pragma unroll
        for (int i = 0; i < 2; ++i)
            #pragma unroll
            for (int j = 0; j < 4; ++j)
                acc[i][j] = wmma_bf16(af[i], wf[j], acc[i][j]);
        __syncthreads();       // all reads of 'buf' done before it is re-filled
        buf = nb;
    }

    #pragma unroll
    for (int j = 0; j < 4; ++j) {
        int n = n0 + wn + j * 16 + rl;
        if (NGUARD && n >= N) continue;
        float bv = BIAS ? bias[n] : 0.0f;
        #pragma unroll
        for (int i = 0; i < 2; ++i) {
            int mb = m0 + wm + i * 16 + half * 8;   // C layout: row = e + 8*half
            #pragma unroll
            for (int e = 0; e < 8; ++e) {
                size_t idx = (size_t)(mb + e) * N + n;
                float v = acc[i][j][e] + bv;
                if (GELU) v = gelu_tanh(v);
                if (RES)  v += res[idx];
                C[idx] = v;
                if (WB16) C16[idx] = f2bf(v);
            }
        }
    }
}

// ---------------- qkv split: fp32 [M,3D] -> bf16 Q(*1/8),K,V [B,H,S,HD] + kv out ----------------
__global__ __launch_bounds__(256) void k_split(const float* __restrict__ qkv,
                                               u16* __restrict__ Qb, u16* __restrict__ Kb,
                                               u16* __restrict__ Vb,
                                               float* __restrict__ kout,
                                               float* __restrict__ vout) {
    size_t i = (size_t)blockIdx.x * 256 + threadIdx.x;
    if (i >= (size_t)MTOK * DMODEL) return;
    int t = (int)(i / DMODEL), d = (int)(i % DMODEL);
    int b = t >> 10, s = t & (SEQ - 1);
    int hh = d >> 6, hd = d & 63;
    size_t dst = ((size_t)(b * HCNT + hh) * SEQ + s) * HDIM + hd;
    size_t src = (size_t)t * (3 * DMODEL) + d;
    float q = qkv[src];
    float k = qkv[src + DMODEL];
    float v = qkv[src + 2 * DMODEL];
    Qb[dst] = f2bf(q * 0.125f);           // 1/sqrt(64) folded into Q
    Kb[dst] = f2bf(k);
    Vb[dst] = f2bf(v);
    kout[dst] = k;
    vout[dst] = v;
}

// ---------------- flash attention (WMMA QK^T and P@V, online softmax) ----------------
// grid: (SEQ/128, B*H); block 256 = 8 waves; each wave owns 16 query rows.
__global__ __launch_bounds__(256) void k_attn(const u16* __restrict__ Qb,
                                              const u16* __restrict__ Kb,
                                              const u16* __restrict__ Vb,
                                              const float* __restrict__ amask,
                                              u16* __restrict__ ctx) {
    __shared__ u16 Vt[64 * LDT];          // V chunk transposed: [hd 64][sk 32]
    __shared__ u16 Ps[8][16 * LDT];       // per-wave P tile: [16 q][32 sk]
    int tid = threadIdx.x, lane = tid & 31, wave = tid >> 5;
    int rl = lane & 15, half = lane >> 4;
    int bh = blockIdx.y;
    int b = bh / HCNT, h = bh % HCNT;
    int q0 = blockIdx.x * 128;
    const u16* Kp = Kb + (size_t)bh * SEQ * HDIM;
    const u16* Vp = Vb + (size_t)bh * SEQ * HDIM;
    const float* msk = amask + (size_t)b * SEQ;

    int q = q0 + wave * 16 + rl;          // A-frag: lane = M row
    const u16* qrow = Qb + ((size_t)bh * SEQ + q) * HDIM;
    v16bf qf0 = load_frag(qrow + half * 8,      qrow + half * 8 + 16);
    v16bf qf1 = load_frag(qrow + 32 + half * 8, qrow + 32 + half * 8 + 16);

    v8f zero = {};
    float mrun[8], lrun[8];
    v8f o[4];
    #pragma unroll
    for (int i = 0; i < 8; ++i) { mrun[i] = -3.402823466e38f; lrun[i] = 0.0f; }
    #pragma unroll
    for (int ns = 0; ns < 4; ++ns) o[ns] = zero;

    for (int kt = 0; kt < SEQ; kt += 32) {
        // stage V[kt..kt+32][0..64] transposed into LDS
        {
            int r = tid >> 3;             // 0..31 (sk)
            int cb = (tid & 7) * 8;       // 0..56 (hd)
            uint4 vv = *(const uint4*)(Vp + (size_t)(kt + r) * HDIM + cb);
            const u16* pv = (const u16*)&vv;
            #pragma unroll
            for (int j = 0; j < 8; ++j) Vt[(cb + j) * LDT + r] = pv[j];
        }
        __syncthreads();

        // scores: two 16-wide key subtiles, K-dim = 64 (2 wmma steps)
        v8f sc[2];
        #pragma unroll
        for (int ns = 0; ns < 2; ++ns) {
            int sk = kt + ns * 16 + rl;   // B-frag: lane = N col (key index)
            const u16* krow = Kp + (size_t)sk * HDIM;
            v16bf kf0 = load_frag(krow + half * 8,      krow + half * 8 + 16);
            v16bf kf1 = load_frag(krow + 32 + half * 8, krow + 32 + half * 8 + 16);
            v8f s = zero;
            s = wmma_bf16(qf0, kf0, s);
            s = wmma_bf16(qf1, kf1, s);
            float am = (1.0f - msk[sk]) * -3.402823466e38f;   // additive mask
            #pragma unroll
            for (int e = 0; e < 8; ++e) s[e] += am;
            sc[ns] = s;
        }

        // online softmax per row (row = i + 8*half), P -> LDS as bf16
        u16* P = &Ps[wave][0];
        #pragma unroll
        for (int i = 0; i < 8; ++i) {
            float a = sc[0][i], c = sc[1][i];
            float t = fmaxf(a, c);
            #pragma unroll
            for (int mds = 1; mds <= 8; mds <<= 1)
                t = fmaxf(t, __shfl_xor(t, mds, 32));
            float mnew = fmaxf(mrun[i], t);
            float esc = __expf(mrun[i] - mnew);
            float p0 = __expf(a - mnew), p1 = __expf(c - mnew);
            float rs = p0 + p1;
            #pragma unroll
            for (int mds = 1; mds <= 8; mds <<= 1)
                rs += __shfl_xor(rs, mds, 32);
            lrun[i] = lrun[i] * esc + rs;
            mrun[i] = mnew;
            #pragma unroll
            for (int ns = 0; ns < 4; ++ns) o[ns][i] *= esc;
            int prow = (i + 8 * half) * LDT;
            P[prow + rl]      = f2bf(p0);
            P[prow + 16 + rl] = f2bf(p1);
        }

        // o += P(16x32) @ V(32x64)
        const u16* pr = P + rl * LDT + half * 8;
        v16bf pf = load_frag(pr, pr + 16);
        #pragma unroll
        for (int ns = 0; ns < 4; ++ns) {
            const u16* vtp = Vt + (ns * 16 + rl) * LDT + half * 8;
            v16bf vf = load_frag(vtp, vtp + 16);
            o[ns] = wmma_bf16(pf, vf, o[ns]);
        }
        __syncthreads();
    }

    // epilogue: ctx[b,s, h*64+hd] (bf16, feeds proj GEMM)
    #pragma unroll
    for (int i = 0; i < 8; ++i) {
        float inv = 1.0f / lrun[i];
        int qq = q0 + wave * 16 + i + 8 * half;
        size_t rowo = ((size_t)(b * SEQ + qq)) * DMODEL + h * HDIM;
        #pragma unroll
        for (int ns = 0; ns < 4; ++ns)
            ctx[rowo + ns * 16 + rl] = f2bf(o[ns][i] * inv);
    }
}

// ---------------- launch ----------------
extern "C" void kernel_launch(void* const* d_in, const int* in_sizes, int n_in,
                              void* d_out, int out_size, void* d_ws, size_t ws_size,
                              hipStream_t stream) {
    (void)in_sizes; (void)n_in; (void)out_size; (void)ws_size;
    const int*   ids    = (const int*)d_in[0];
    const float* amask  = (const float*)d_in[1];
    const int*   cp     = (const int*)d_in[2];
    // d_in[3] = past_kv (all zeros, fully overwritten) -- unused
    const float* wte    = (const float*)d_in[4];
    const float* wpe    = (const float*)d_in[5];
    const float* ln1_g  = (const float*)d_in[6];
    const float* ln1_b  = (const float*)d_in[7];
    const float* attn_w = (const float*)d_in[8];
    const float* attn_b = (const float*)d_in[9];
    const float* proj_w = (const float*)d_in[10];
    const float* proj_b = (const float*)d_in[11];
    const float* ln2_g  = (const float*)d_in[12];
    const float* ln2_b  = (const float*)d_in[13];
    const float* fc_w   = (const float*)d_in[14];
    const float* fc_b   = (const float*)d_in[15];
    const float* mlp_w  = (const float*)d_in[16];
    const float* mlp_b  = (const float*)d_in[17];
    const float* lnf_g  = (const float*)d_in[18];
    const float* lnf_b  = (const float*)d_in[19];
    float* out = (float*)d_out;

    // workspace bump allocator (256B aligned)
    char* wsp = (char*)d_ws;
    auto alloc = [&](size_t bytes) -> void* {
        void* p = (void*)wsp;
        wsp += (bytes + 255) & ~(size_t)255;
        return p;
    };
    u16* wqkvT = (u16*)alloc((size_t)LCNT * 3 * DMODEL * DMODEL * 2);
    u16* wprojT= (u16*)alloc((size_t)LCNT * DMODEL * DMODEL * 2);
    u16* wfcT  = (u16*)alloc((size_t)LCNT * FFDIM * DMODEL * 2);
    u16* wmlpT = (u16*)alloc((size_t)LCNT * DMODEL * FFDIM * 2);
    u16* wteb  = (u16*)alloc((size_t)VOCAB * DMODEL * 2);
    u16* lnb   = (u16*)alloc((size_t)MTOK * DMODEL * 2);
    u16* ctxb  = (u16*)alloc((size_t)MTOK * DMODEL * 2);
    u16* ffb   = (u16*)alloc((size_t)MTOK * FFDIM * 2);
    u16* Qb    = (u16*)alloc(KVL * 2);
    u16* Kb    = (u16*)alloc(KVL * 2);
    u16* Vb    = (u16*)alloc(KVL * 2);
    float* hbuf= (float*)alloc((size_t)MTOK * DMODEL * 4);
    float* qkvb= (float*)alloc((size_t)MTOK * 3 * DMODEL * 4);
    float* ffs = (float*)alloc((size_t)MTOK * FFDIM * 4);

    dim3 tb(32, 8);
    // weight conversion (deterministic, every call)
    k_tconv<<<dim3(3 * DMODEL / 32, DMODEL / 32, LCNT), tb, 0, stream>>>(attn_w, wqkvT, DMODEL, 3 * DMODEL);
    k_tconv<<<dim3(DMODEL / 32, DMODEL / 32, LCNT),     tb, 0, stream>>>(proj_w, wprojT, DMODEL, DMODEL);
    k_tconv<<<dim3(FFDIM / 32, DMODEL / 32, LCNT),      tb, 0, stream>>>(fc_w,   wfcT,   DMODEL, FFDIM);
    k_tconv<<<dim3(DMODEL / 32, FFDIM / 32, LCNT),      tb, 0, stream>>>(mlp_w,  wmlpT,  FFDIM,  DMODEL);
    k_conv<<<4096, 256, 0, stream>>>(wte, wteb, (size_t)VOCAB * DMODEL);

    // embedding
    k_embed<<<(MTOK * DMODEL) / 256, 256, 0, stream>>>(ids, cp, wte, wpe, hbuf);

    for (int l = 0; l < LCNT; ++l) {
        float* kv_k = out + LOGITS_ELEMS + (size_t)l * 2 * KVL;
        float* kv_v = kv_k + KVL;

        k_ln<<<MTOK, 256, 0, stream>>>(hbuf, ln1_g + l * DMODEL, ln1_b + l * DMODEL, lnb);
        // qkv = ln1 @ Wqkv + b
        k_gemm<false, false, false, true, false><<<dim3(3 * DMODEL / 128, MTOK / 128), 256, 0, stream>>>(
            lnb, wqkvT + (size_t)l * 3 * DMODEL * DMODEL, attn_b + (size_t)l * 3 * DMODEL,
            nullptr, qkvb, nullptr, MTOK, 3 * DMODEL, DMODEL);
        k_split<<<(MTOK * DMODEL) / 256, 256, 0, stream>>>(qkvb, Qb, Kb, Vb, kv_k, kv_v);
        k_attn<<<dim3(SEQ / 128, BATCH * HCNT), 256, 0, stream>>>(Qb, Kb, Vb, amask, ctxb);
        // h += ctx @ Wproj + b
        k_gemm<false, true, false, true, false><<<dim3(DMODEL / 128, MTOK / 128), 256, 0, stream>>>(
            ctxb, wprojT + (size_t)l * DMODEL * DMODEL, proj_b + (size_t)l * DMODEL,
            hbuf, hbuf, nullptr, MTOK, DMODEL, DMODEL);
        k_ln<<<MTOK, 256, 0, stream>>>(hbuf, ln2_g + l * DMODEL, ln2_b + l * DMODEL, lnb);
        // ff = gelu(ln2 @ Wfc + b), bf16 copy for next GEMM
        k_gemm<true, false, true, true, false><<<dim3(FFDIM / 128, MTOK / 128), 256, 0, stream>>>(
            lnb, wfcT + (size_t)l * FFDIM * DMODEL, fc_b + (size_t)l * FFDIM,
            nullptr, ffs, ffb, MTOK, FFDIM, DMODEL);
        // h += ff @ Wmlp + b
        k_gemm<false, true, false, true, false><<<dim3(DMODEL / 128, MTOK / 128), 256, 0, stream>>>(
            ffb, wmlpT + (size_t)l * DMODEL * FFDIM, mlp_b + (size_t)l * DMODEL,
            hbuf, hbuf, nullptr, MTOK, DMODEL, FFDIM);
    }

    k_ln<<<MTOK, 256, 0, stream>>>(hbuf, lnf_g, lnf_b, lnb);
    // logits = lnf @ wte^T   (N = 50257 -> bounds-guarded tiles)
    k_gemm<false, false, false, false, true><<<dim3((VOCAB + 127) / 128, MTOK / 128), 256, 0, stream>>>(
        lnb, wteb, nullptr, nullptr, out, nullptr, MTOK, VOCAB, DMODEL);
}
